// PagedLlamaModelV1_19361712571224
// MI455X (gfx1250) — matrix operand order repaired and checked by reading, not compiled, not run on hardware
//
#include <hip/hip_runtime.h>
#include <math.h>

// ---------------- problem constants ----------------
#define BS  2
#define SL  2048
#define H   32
#define DH  128
#define STRIDE_T 16
#define NUM_PAGES (BS * (SL / STRIDE_T))          // 256
#define PAGE_FLOATS (2 * STRIDE_T * H * DH)       // 131072 floats per page
#define HALF_PAGE (STRIDE_T * H * DH)             // 65536 floats (K or V partition)

#define KT    64            // keys per iteration (4 x 16 score sub-tiles)
#define LDK   136           // row stride (halfs) of K tile   [64][136]
#define LDVT  72            // row stride (halfs) of Vt tile  [128][72]
#define LDP   72            // row stride (halfs) of P tile   [16][72]
#define LDQ   136           // row stride (halfs) of Q stage  [16][136]

typedef __attribute__((ext_vector_type(16))) _Float16 v16h;
typedef __attribute__((ext_vector_type(8)))  _Float16 v8h;
typedef __attribute__((ext_vector_type(4)))  _Float16 v4h;
typedef __attribute__((ext_vector_type(8)))  float    v8f;

__device__ inline v8f wmma_f16(v16h a, v16h b, v8f c) {
  return __builtin_amdgcn_wmma_f32_16x16x32_f16(
      /*neg_a=*/false, a, /*neg_b=*/false, b,
      /*c_mod=*/(short)0, c, /*reuse_a=*/false, /*reuse_b=*/false);
}

// build a v16h fragment from two 16-byte LDS chunks
__device__ inline v16h lds_frag(const _Float16* p0, const _Float16* p1) {
  v16h r;
  ((v8h*)&r)[0] = *(const v8h*)p0;
  ((v8h*)&r)[1] = *(const v8h*)p1;
  return r;
}

// A-operand fragment (16x32 f16) from row-major LDS tile, rows = M
__device__ inline v16h ldsA(const _Float16* tile, int ld, int k0, int lane) {
  const int m    = lane & 15;
  const int koff = (lane >> 4) << 3;        // 0 or 8
  const _Float16* p = tile + m * ld + k0 + koff;
  return lds_frag(p, p + 16);
}

// B-operand fragment (32x16 f16): LDS tile holds B^T row-major, row = N
__device__ inline v16h ldsB(const _Float16* tile, int ld, int n0, int k0, int lane) {
  const int n  = lane & 15;
  const int kb = (lane >> 4) << 3;          // 0 or 8
  const _Float16* p = tile + (n0 + n) * ld + k0 + kb;
  return lds_frag(p, p + 16);
}

__device__ inline float rmax16(float x) {
  x = fmaxf(x, __shfl_xor(x, 1, 32));
  x = fmaxf(x, __shfl_xor(x, 2, 32));
  x = fmaxf(x, __shfl_xor(x, 4, 32));
  x = fmaxf(x, __shfl_xor(x, 8, 32));
  return x;
}
__device__ inline float rsum16(float x) {
  x += __shfl_xor(x, 1, 32);
  x += __shfl_xor(x, 2, 32);
  x += __shfl_xor(x, 4, 32);
  x += __shfl_xor(x, 8, 32);
  return x;
}

// ---------------- paged cache write ----------------
__global__ void cache_write_kernel(const float* __restrict__ xk,
                                   const float* __restrict__ xv,
                                   const int* __restrict__ pids,
                                   float* __restrict__ cache_out) {
  const int i   = blockIdx.x;                 // logical block 0..255
  const int pid = pids[i];
  const float4* srcK = (const float4*)(xk + (size_t)i * HALF_PAGE);
  const float4* srcV = (const float4*)(xv + (size_t)i * HALF_PAGE);
  float4* dstK = (float4*)(cache_out + (size_t)pid * PAGE_FLOATS);
  float4* dstV = dstK + (HALF_PAGE / 4);
  for (int t = threadIdx.x; t < HALF_PAGE / 4; t += blockDim.x) {
    dstK[t] = srcK[t];
    dstV[t] = srcV[t];
  }
}

// ---------------- flash attention (f16 WMMA, f32 accum, exp2 softmax) ----------------
__global__ __launch_bounds__(128)
void flash_attn_kernel(const float* __restrict__ xq,
                       const float* __restrict__ xk,
                       const float* __restrict__ xv,
                       float* __restrict__ out) {
  __shared__ __align__(16) _Float16 sK[KT * LDK];        // 17408 B
  __shared__ __align__(16) _Float16 sVt[DH * LDVT];      // 18432 B
  __shared__ __align__(16) _Float16 sP[4 * 16 * LDP];    //  9216 B
  __shared__ __align__(16) _Float16 sQ[4 * 16 * LDQ];    // 17408 B

  const int tid  = threadIdx.x;
  const int w    = tid >> 5;
  const int lane = tid & 31;
  const int col  = lane & 15;
  const int rowoff = (lane >> 4) << 3;      // 0 or 8 (C/D layout row group)

  const int NQB = SL / 64;                  // 32 query super-tiles per (b,h)
  const int qb  = blockIdx.x % NQB;
  const int bh  = blockIdx.x / NQB;
  const int h   = bh % H;
  const int bidx = bh / H;
  const int q0  = qb * 64 + w * 16;         // this wave's query tile base

  // fold softmax scale AND log2(e) into Q so scores are exp2-ready:
  // qscale = (1/sqrt(128)) * log2(e)
  const float qscale = 0.12751741f;

  // ---- stage this wave's Q tile (16x128) into LDS as f16, build A fragments
  _Float16* qs = sQ + w * 16 * LDQ;
  {
    #pragma unroll
    for (int i = 0; i < 16; ++i) {
      int idx4 = lane + i * 32;             // 512 float4 per tile
      int row  = idx4 >> 5;
      int c4   = idx4 & 31;
      const float4 f = *(const float4*)(xq +
          ((size_t)(bidx * SL + q0 + row) * H + h) * DH + c4 * 4);
      v4h hv = { (_Float16)(f.x * qscale), (_Float16)(f.y * qscale),
                 (_Float16)(f.z * qscale), (_Float16)(f.w * qscale) };
      *(v4h*)(qs + row * LDQ + c4 * 4) = hv;
    }
  }
  asm volatile("s_wait_dscnt 0" ::: "memory");
  v16h qf[4];
  #pragma unroll
  for (int j = 0; j < 4; ++j) qf[j] = ldsA(qs, LDQ, j * 32, lane);

  // ---- running flash-attention state (exp2 domain)
  v8f acc[8];
  #pragma unroll
  for (int n = 0; n < 8; ++n) acc[n] = (v8f){0.f,0.f,0.f,0.f,0.f,0.f,0.f,0.f};
  float mrun[8], lrun[8];
  #pragma unroll
  for (int g = 0; g < 8; ++g) { mrun[g] = -INFINITY; lrun[g] = 0.f; }

  _Float16* pw = sP + w * 16 * LDP;

  for (int kb = 0; kb < SL; kb += KT) {
    __syncthreads();   // previous iteration's K/V reads complete

    // cooperative K/V tile load (128 threads, float4, f32 -> f16)
    #pragma unroll
    for (int i = 0; i < 16; ++i) {
      int idx4 = tid + i * 128;             // 2048 float4 per tensor tile
      int row  = idx4 >> 5;                 // key within tile (0..63)
      int c4   = idx4 & 31;
      const size_t gbase =
          ((size_t)(bidx * SL + kb + row) * H + h) * DH + c4 * 4;
      const float4 fk = *(const float4*)(xk + gbase);
      const float4 fv = *(const float4*)(xv + gbase);
      v4h hk = { (_Float16)fk.x, (_Float16)fk.y, (_Float16)fk.z, (_Float16)fk.w };
      *(v4h*)(sK + row * LDK + c4 * 4) = hk;
      int d = c4 * 4;                       // transposed V store
      sVt[(d + 0) * LDVT + row] = (_Float16)fv.x;
      sVt[(d + 1) * LDVT + row] = (_Float16)fv.y;
      sVt[(d + 2) * LDVT + row] = (_Float16)fv.z;
      sVt[(d + 3) * LDVT + row] = (_Float16)fv.w;
    }
    if (kb + KT < SL) {   // prefetch next key tile (global_prefetch_b8)
      const size_t nb = ((size_t)(bidx * SL + kb + KT) * H + h) * DH;
      __builtin_prefetch(xk + nb, 0, 3);
      __builtin_prefetch(xv + nb, 0, 3);
    }
    __syncthreads();

    // ---- scores: S[16x64] = Q(16x128) . K^T(128x64), four 16x16 tiles.
    // Preload 8 B-fragments at a time so WMMAs issue back-to-back.
    v8f s[4];
    #pragma unroll
    for (int t = 0; t < 4; ++t) s[t] = (v8f){0.f,0.f,0.f,0.f,0.f,0.f,0.f,0.f};
    #pragma unroll
    for (int hlf = 0; hlf < 2; ++hlf) {
      v16h bk[8];
      #pragma unroll
      for (int j = 0; j < 4; ++j) {
        bk[j]     = ldsB(sK, LDK, (2 * hlf + 0) * 16, j * 32, lane);
        bk[4 + j] = ldsB(sK, LDK, (2 * hlf + 1) * 16, j * 32, lane);
      }
      #pragma unroll
      for (int j = 0; j < 4; ++j) s[2*hlf+0] = wmma_f16(qf[j], bk[j],     s[2*hlf+0]);
      #pragma unroll
      for (int j = 0; j < 4; ++j) s[2*hlf+1] = wmma_f16(qf[j], bk[4 + j], s[2*hlf+1]);
    }

    // ---- online softmax update in exp2 domain (rows in 16-lane groups)
    float alpha[8];
    #pragma unroll
    for (int g = 0; g < 8; ++g) {
      float a0 = s[0][g], a1 = s[1][g], a2 = s[2][g], a3 = s[3][g];
      float mx = rmax16(fmaxf(fmaxf(a0, a1), fmaxf(a2, a3)));
      float mn = fmaxf(mrun[g], mx);
      float al = __builtin_amdgcn_exp2f(mrun[g] - mn);
      float p0 = __builtin_amdgcn_exp2f(a0 - mn);
      float p1 = __builtin_amdgcn_exp2f(a1 - mn);
      float p2 = __builtin_amdgcn_exp2f(a2 - mn);
      float p3 = __builtin_amdgcn_exp2f(a3 - mn);
      lrun[g]  = lrun[g] * al + rsum16(((p0 + p1) + (p2 + p3)));
      mrun[g]  = mn;
      alpha[g] = al;
      const int prow = (g + rowoff) * LDP + col;
      pw[prow]      = (_Float16)p0;
      pw[prow + 16] = (_Float16)p1;
      pw[prow + 32] = (_Float16)p2;
      pw[prow + 48] = (_Float16)p3;
    }
    asm volatile("s_wait_dscnt 0" ::: "memory");
    v16h pf0 = ldsA(pw, LDP, 0,  lane);
    v16h pf1 = ldsA(pw, LDP, 32, lane);

    // ---- out += P(16x64) . V(64x128): 2 K-chunks x 8 N-tiles.
    // Rescale accumulators while the first chunk's V fragments are in flight.
    #pragma unroll
    for (int c = 0; c < 2; ++c) {
      v16h vfr[8];
      #pragma unroll
      for (int n = 0; n < 8; ++n) vfr[n] = ldsB(sVt, LDVT, n * 16, c * 32, lane);
      if (c == 0) {
        #pragma unroll
        for (int n = 0; n < 8; ++n)
          #pragma unroll
          for (int g = 0; g < 8; ++g) acc[n][g] *= alpha[g];
      }
      v16h pf = c ? pf1 : pf0;
      #pragma unroll
      for (int n = 0; n < 8; ++n) acc[n] = wmma_f16(pf, vfr[n], acc[n]);
    }
  }

  // ---- normalize and store output [BS, SL, H*DH]
  float inv[8];
  #pragma unroll
  for (int g = 0; g < 8; ++g) inv[g] = 1.0f / lrun[g];
  #pragma unroll
  for (int n = 0; n < 8; ++n) {
    #pragma unroll
    for (int g = 0; g < 8; ++g) {
      int row = g + rowoff;
      int q   = q0 + row;
      int d   = n * 16 + col;
      out[((size_t)(bidx * SL + q) * H + h) * DH + d] = acc[n][g] * inv[g];
    }
  }
}

extern "C" void kernel_launch(void* const* d_in, const int* in_sizes, int n_in,
                              void* d_out, int out_size, void* d_ws, size_t ws_size,
                              hipStream_t stream) {
  const float* xq   = (const float*)d_in[0];
  const float* xk   = (const float*)d_in[1];
  const float* xv   = (const float*)d_in[2];
  const int*   pids = (const int*)d_in[4];   // seq_block_ids (int32)

  float* attn_out  = (float*)d_out;                                  // [BS,SL,H*DH]
  float* cache_out = attn_out + (size_t)BS * SL * H * DH;            // [NUM_PAGES,PAGE_FLOATS]

  // 1) paged cache write (scatter copy; all pages covered by arange ids)
  cache_write_kernel<<<NUM_PAGES, 256, 0, stream>>>(xk, xv, pids, cache_out);

  // 2) flash attention: (BS*H) heads x (SL/64) query super-tiles, 4 waves each
  const int grid = BS * H * (SL / 64);  // 2048
  flash_attn_kernel<<<grid, 128, 0, stream>>>(xq, xk, xv, attn_out);
}
